// DeepSet_39805756899872
// MI455X (gfx1250) — matrix operand order repaired and checked by reading, not compiled, not run on hardware
//
#include <hip/hip_runtime.h>
#include <hip/hip_bf16.h>

// ---------------- constants from the reference ----------------
#define N_ATOMS   50000
#define N_EDGES   400000
#define EMB       256
#define NUM_RBF   50
#define NUM_GATES 10
#define F_OUT     128
#define OUTER_CUTOFF 5.0f

// output layout (floats / 32-bit words)
#define OUT_ATOM   0                       // 50000*128
#define OUT_VEC    6400000                 // 50000*3*128
#define OUT_Z      25600000                // 50000 (int bits)
#define OUT_POS    25650000                // 150000
#define OUT_BATCH  25800000                // 50000 (int bits)

// workspace layout (bf16 elements; every matrix stored transposed [dout][din_pad])
#define WT_F0   0            // [256][64]   fused W_dist @ W_dt[0:256] (K pad 50->64)
#define WT_F1   16384        // [256][256]  fused W_ai @ W_gamma[0:256]
#define WT_F2   81920        // [256][256]  fused W_aj @ W_gamma[256:512]
#define WT_G3   147456       // [256][256]  W_gamma[512:768]
#define WT_EXP  212992       // [10][128][256]
#define WS_F32  540672       // ushort offset of fp32 region: bvec[256], beta[256]

typedef __attribute__((ext_vector_type(16))) __bf16        v16bf;
typedef __attribute__((ext_vector_type(8)))  float         v8f;
typedef __attribute__((ext_vector_type(8)))  unsigned int  v8u;
typedef __attribute__((ext_vector_type(4)))  unsigned int  v4u;

__device__ __forceinline__ unsigned short f2bf(float f) {
  unsigned u = __builtin_bit_cast(unsigned, f);
  unsigned rnd = 0x7FFFu + ((u >> 16) & 1u);
  return (unsigned short)((u + rnd) >> 16);
}

// ---------------- prep: plain transpose fp32 [din][dout] -> bf16 [dout][din_pad]
__global__ void transpose_bf16(const float* __restrict__ W,
                               unsigned short* __restrict__ Wt,
                               int din, int dout, int din_pad) {
  int idx = blockIdx.x * blockDim.x + threadIdx.x;
  if (idx >= dout * din_pad) return;
  int j = idx / din_pad;
  int i = idx % din_pad;
  float v = (i < din) ? W[i * dout + j] : 0.0f;
  Wt[idx] = f2bf(v);
}

// ---------------- prep: fused product  (A[din][K] @ B[K][dout]) -> bf16 [dout][din_pad]
__global__ void fuse_t_bf16(const float* __restrict__ A,
                            const float* __restrict__ B,
                            unsigned short* __restrict__ Wt,
                            int din, int K, int dout, int din_pad) {
  int idx = blockIdx.x * blockDim.x + threadIdx.x;
  if (idx >= dout * din_pad) return;
  int j = idx / din_pad;
  int i = idx % din_pad;
  float s = 0.0f;
  if (i < din)
    for (int k = 0; k < K; ++k) s += A[i * K + k] * B[k * dout + j];
  Wt[idx] = f2bf(s);
}

// ---------------- prep: fused bias vectors (fp32, exact) ----------------
// bvec[n] = b_dist @ W_dt[0:256]   ;   beta[n] = b_ai@G1 + b_aj@G2 + b_gamma
__global__ void bias_prep(const float* __restrict__ b_dist, const float* __restrict__ W_dt,
                          const float* __restrict__ b_ai, const float* __restrict__ b_aj,
                          const float* __restrict__ b_gamma, const float* __restrict__ W_gamma,
                          float* __restrict__ bvec, float* __restrict__ beta) {
  int n = blockIdx.x * blockDim.x + threadIdx.x;
  if (n >= 256) return;
  float s0 = 0.0f, s1 = b_gamma[n];
  for (int k = 0; k < 256; ++k) {
    s0 += b_dist[k] * W_dt[k * 256 + n];
    s1 += b_ai[k] * W_gamma[k * 256 + n] + b_aj[k] * W_gamma[(256 + k) * 256 + n];
  }
  bvec[n] = s0;
  beta[n] = s1;
}

// ---------------- zero accumulators + passthrough copies -------------------
__global__ void init_out_kernel(float* __restrict__ out,
                                const unsigned* __restrict__ z,
                                const float* __restrict__ pos,
                                const unsigned* __restrict__ batch) {
  unsigned i = blockIdx.x * blockDim.x + threadIdx.x;
  if (i < 25600000u) out[i] = 0.0f;
  if (i < 50000u)  ((unsigned*)out)[OUT_Z + i] = z[i];
  if (i < 150000u) out[OUT_POS + i] = pos[i];
  if (i < 50000u)  ((unsigned*)out)[OUT_BATCH + i] = batch[i];
}

// ---------------- bf16 WMMA tile GEMM ------------------------------------
// A tile in LDS row-major [16][astride] bf16; B in ws transposed [dout][din].
// A fragment = two contiguous 16B chunks (ds_load_b128 x2); B = 32B contiguous.
template <int NT, int KS>
__device__ __forceinline__ void gemm_bf16(const unsigned short* As, int astride,
                                          const unsigned short* __restrict__ Wt,
                                          int din, int ntoff, v8f* acc, int lane) {
  const int m  = lane & 15;
  const int hi = lane >> 4;
#pragma unroll
  for (int ks = 0; ks < KS; ++ks) {
    const unsigned short* row = As + m * astride + ks * 32 + hi * 16;
    v4u a0 = *(const v4u*)(row);        // K 0..7   (or 16..23 for hi lanes)
    v4u a1 = *(const v4u*)(row + 8);    // K 8..15  (or 24..31)
    v8u au;
#pragma unroll
    for (int i = 0; i < 4; ++i) { au[i] = a0[i]; au[4 + i] = a1[i]; }
    v16bf a = __builtin_bit_cast(v16bf, au);
#pragma unroll
    for (int nt = 0; nt < NT; ++nt) {
      const v8u* bp = (const v8u*)(Wt + ((ntoff + nt) * 16 + m) * din + ks * 32 + hi * 16);
      v16bf b = __builtin_bit_cast(v16bf, *bp);
      acc[nt] = __builtin_amdgcn_wmma_f32_16x16x32_bf16(
          false, a, false, b, (short)0, acc[nt], false, false);
    }
  }
}

// ---------------- main fused edge kernel -----------------------------------
__global__ __launch_bounds__(64, 1)
void edge_moe_kernel(const int* __restrict__ edge_index,
                     const float* __restrict__ edge_vec,
                     const int* __restrict__ z,
                     const float* __restrict__ emb_table,
                     const float* __restrict__ W_dt,      // fp32, tail rows 256..261
                     const float* __restrict__ b_dt,
                     const float* __restrict__ b_exp,
                     const float* __restrict__ t_params,
                     const unsigned short* __restrict__ ws,
                     const float* __restrict__ bvec,
                     const float* __restrict__ beta,
                     float* __restrict__ atom_out,
                     float* __restrict__ vec_out) {
  __shared__ unsigned short bufP[2][16 * 256];
  __shared__ unsigned short bufQ[2][16 * 256];
  __shared__ unsigned short bufR[2][16 * 256];
  __shared__ float s_C[2][16];
  __shared__ float s_sf[2][16][6];
  __shared__ float s_gate[2][16][NUM_GATES];
  __shared__ float s_ev[2][16][3];
  __shared__ int   s_src[2][16];
  __shared__ int   s_zsrc[2][16];
  __shared__ int   s_zdst[2][16];

  const int lane = threadIdx.x & 31;
  const int w    = threadIdx.x >> 5;
  const int tile = blockIdx.x * 2 + w;        // 25000 tiles, exact
  const int l15  = lane & 15;
  const int hi   = lane >> 4;

  // ---- phase 0: per-edge scalars (lanes 0..15) + packed RBF tile -> P ----
  if (lane < 16) {
    int m = lane;
    int e = tile * 16 + m;
    int src = edge_index[e];
    int dst = edge_index[N_EDGES + e];
    s_src[w][m]  = src;
    s_zsrc[w][m] = z[src];
    s_zdst[w][m] = z[dst];
    float vx = edge_vec[e * 3 + 0];
    float vy = edge_vec[e * 3 + 1];
    float vz = edge_vec[e * 3 + 2];
    float d  = sqrtf(vx * vx + vy * vy + vz * vz);
    float inv = 1.0f / d;
    s_ev[w][m][0] = vx * inv; s_ev[w][m][1] = vy * inv; s_ev[w][m][2] = vz * inv;
    s_C[w][m] = (d < OUTER_CUTOFF)
                    ? 0.5f * (__cosf(d * 3.14159265358979f / OUTER_CUTOFF) + 1.0f)
                    : 0.0f;
    float d2 = d * d;
    s_sf[w][m][0] = d2;
    s_sf[w][m][1] = d2 * d;
    s_sf[w][m][2] = d2 * d2;
    s_sf[w][m][3] = sqrtf(d);
    s_sf[w][m][4] = __logf(d);
    s_sf[w][m][5] = d;
    // gate = softmax over 1/max(|d-t|,1e-8)
    float wv[NUM_GATES], mx = -1e30f;
#pragma unroll
    for (int g = 0; g < NUM_GATES; ++g) {
      float a = fabsf(d - t_params[g]);
      wv[g] = 1.0f / fmaxf(a, 1e-8f);
      mx = fmaxf(mx, wv[g]);
    }
    float ssum = 0.0f;
#pragma unroll
    for (int g = 0; g < NUM_GATES; ++g) { wv[g] = __expf(wv[g] - mx); ssum += wv[g]; }
    float rs = 1.0f / ssum;
#pragma unroll
    for (int g = 0; g < NUM_GATES; ++g) s_gate[w][m][g] = wv[g] * rs;
    // RBF row, K padded to 64, two bf16 packed per b32 store
    const float step  = OUTER_CUTOFF / (NUM_RBF - 1);
    const float coeff = -0.5f / (step * step);
    for (int r2 = 0; r2 < 32; ++r2) {
      int r = r2 * 2;
      float dd0 = d - r * step, dd1 = d - (r + 1) * step;
      float v0 = (r     < NUM_RBF) ? __expf(coeff * dd0 * dd0) : 0.0f;
      float v1 = (r + 1 < NUM_RBF) ? __expf(coeff * dd1 * dd1) : 0.0f;
      ((unsigned*)bufP[w])[m * 32 + r2] =
          (unsigned)f2bf(v0) | ((unsigned)f2bf(v1) << 16);
    }
  }
  __syncthreads();

  // ---- gather x[dst] -> R (bf16, packed pairs) ----
  for (int i = lane; i < 16 * 128; i += 32) {
    int row = i >> 7;
    int c   = (i & 127) << 1;
    const float* p = emb_table + s_zdst[w][row] * EMB + c;
    ((unsigned*)bufR[w])[(row << 7) + (i & 127)] =
        (unsigned)f2bf(p[0]) | ((unsigned)f2bf(p[1]) << 16);
  }

  // ---- d_t = C (.) (rbf @ Wf0 + bvec) + sf @ W_dt_tail + b_dt  -> Q ------
  {
    float cl[8], sfl[8][6];
#pragma unroll
    for (int r = 0; r < 8; ++r) {
      int mm = r + 8 * hi;
      cl[r] = s_C[w][mm];
#pragma unroll
      for (int f = 0; f < 6; ++f) sfl[r][f] = s_sf[w][mm][f];
    }
#pragma unroll 1
    for (int half = 0; half < 2; ++half) {
      v8f acc[8] = {};
      gemm_bf16<8, 2>(bufP[w], 64, ws + WT_F0, 64, half * 8, acc, lane);
#pragma unroll
      for (int nt = 0; nt < 8; ++nt) {
        int nn = (half * 8 + nt) * 16 + l15;
        float bv = bvec[nn];
        float bd = b_dt[nn];
        float wt[6];
#pragma unroll
        for (int f = 0; f < 6; ++f) wt[f] = W_dt[(256 + f) * 256 + nn];
#pragma unroll
        for (int r = 0; r < 8; ++r) {
          float v = cl[r] * (acc[nt][r] + bv) + bd;
#pragma unroll
          for (int f = 0; f < 6; ++f) v += sfl[r][f] * wt[f];
          bufQ[w][(r + 8 * hi) * 256 + nn] = f2bf(v);
        }
      }
    }
  }
  __syncthreads();

  // ---- gather x[src] -> P (RBF tile dead) ----
  for (int i = lane; i < 16 * 128; i += 32) {
    int row = i >> 7;
    int c   = (i & 127) << 1;
    const float* p = emb_table + s_zsrc[w][row] * EMB + c;
    ((unsigned*)bufP[w])[(row << 7) + (i & 127)] =
        (unsigned)f2bf(p[0]) | ((unsigned)f2bf(p[1]) << 16);
  }
  __syncthreads();

  // ---- gamma = x_src@F1 + x_dst@F2 + d_t@G3 + beta -> P (in place) -------
  {
    v8f acc[16] = {};
    gemm_bf16<16, 8>(bufP[w], 256, ws + WT_F1, 256, 0, acc, lane);
    gemm_bf16<16, 8>(bufR[w], 256, ws + WT_F2, 256, 0, acc, lane);
    gemm_bf16<16, 8>(bufQ[w], 256, ws + WT_G3, 256, 0, acc, lane);
#pragma unroll
    for (int nt = 0; nt < 16; ++nt) {
      int nn = nt * 16 + l15;
      float bb = beta[nn];
#pragma unroll
      for (int r = 0; r < 8; ++r)
        bufP[w][(r + 8 * hi) * 256 + nn] = f2bf(acc[nt][r] + bb);
    }
  }
  __syncthreads();

  // ---- experts: edge_out = sum_g gate_g * (gamma @ W_exp[g] + b_exp[g]) ---
  v8f eo[8] = {};
#pragma unroll 1
  for (int g = 0; g < NUM_GATES; ++g) {
    v8f pacc[8] = {};
    gemm_bf16<8, 8>(bufP[w], 256, ws + WT_EXP + g * (F_OUT * 256), 256, 0, pacc, lane);
    float gl[8];
#pragma unroll
    for (int r = 0; r < 8; ++r) gl[r] = s_gate[w][r + 8 * hi][g];
#pragma unroll
    for (int nt = 0; nt < 8; ++nt) {
      float bb = b_exp[g * F_OUT + nt * 16 + l15];
#pragma unroll
      for (int r = 0; r < 8; ++r) eo[nt][r] += gl[r] * (pacc[nt][r] + bb);
    }
  }

  // ---- scatter: atom_out += edge_out ; vec += ev (x) edge_out -------------
  {
    int   atm[8];
    float e0[8], e1[8], e2[8];
#pragma unroll
    for (int r = 0; r < 8; ++r) {
      int mm = r + 8 * hi;
      atm[r] = s_src[w][mm];
      e0[r] = s_ev[w][mm][0]; e1[r] = s_ev[w][mm][1]; e2[r] = s_ev[w][mm][2];
    }
#pragma unroll
    for (int nt = 0; nt < 8; ++nt) {
      int nn = nt * 16 + l15;
#pragma unroll
      for (int r = 0; r < 8; ++r) {
        float v = eo[nt][r];
        unsafeAtomicAdd(&atom_out[atm[r] * F_OUT + nn], v);
        unsafeAtomicAdd(&vec_out[atm[r] * (3 * F_OUT) + nn], e0[r] * v);
        unsafeAtomicAdd(&vec_out[atm[r] * (3 * F_OUT) + F_OUT + nn], e1[r] * v);
        unsafeAtomicAdd(&vec_out[atm[r] * (3 * F_OUT) + 2 * F_OUT + nn], e2[r] * v);
      }
    }
  }
}

// ---------------- host-side launcher ---------------------------------------
extern "C" void kernel_launch(void* const* d_in, const int* in_sizes, int n_in,
                              void* d_out, int out_size, void* d_ws, size_t ws_size,
                              hipStream_t stream) {
  (void)in_sizes; (void)n_in; (void)out_size; (void)ws_size;
  const int*   z        = (const int*)d_in[0];
  const float* pos      = (const float*)d_in[1];
  const int*   batch    = (const int*)d_in[2];
  const int*   edge_idx = (const int*)d_in[3];
  const float* edge_vec = (const float*)d_in[4];
  const float* emb      = (const float*)d_in[5];
  const float* W_dist   = (const float*)d_in[6];
  const float* b_dist   = (const float*)d_in[7];
  const float* W_dt     = (const float*)d_in[8];
  const float* b_dt     = (const float*)d_in[9];
  const float* W_ai     = (const float*)d_in[10];
  const float* b_ai     = (const float*)d_in[11];
  const float* W_aj     = (const float*)d_in[12];
  const float* b_aj     = (const float*)d_in[13];
  const float* W_gamma  = (const float*)d_in[14];
  const float* b_gamma  = (const float*)d_in[15];
  const float* W_exp    = (const float*)d_in[16];
  const float* b_exp    = (const float*)d_in[17];
  const float* t_params = (const float*)d_in[18];
  unsigned short* ws = (unsigned short*)d_ws;
  float* fws = (float*)(ws + WS_F32);       // bvec[256], beta[256]
  float* out = (float*)d_out;

  // fused weight prep (bf16 transposed, L2-resident, ~1.1 MB)
  fuse_t_bf16<<<(256 * 64 + 255) / 256, 256, 0, stream>>>(
      W_dist, W_dt, ws + WT_F0, 50, 256, 256, 64);              // W_dist @ W_dt0
  fuse_t_bf16<<<256, 256, 0, stream>>>(
      W_ai, W_gamma, ws + WT_F1, 256, 256, 256, 256);           // W_ai @ G1
  fuse_t_bf16<<<256, 256, 0, stream>>>(
      W_aj, W_gamma + 256 * 256, ws + WT_F2, 256, 256, 256, 256); // W_aj @ G2
  transpose_bf16<<<256, 256, 0, stream>>>(
      W_gamma + 2 * 256 * 256, ws + WT_G3, 256, 256, 256);      // G3
  for (int g = 0; g < NUM_GATES; ++g)
    transpose_bf16<<<128, 256, 0, stream>>>(W_exp + g * (256 * F_OUT),
                                            ws + WT_EXP + g * (F_OUT * 256),
                                            256, F_OUT, 256);
  bias_prep<<<1, 256, 0, stream>>>(b_dist, W_dt, b_ai, b_aj, b_gamma, W_gamma,
                                   fws, fws + 256);

  // zero accumulators, copy z/pos/batch passthrough outputs
  init_out_kernel<<<100000, 256, 0, stream>>>(out, (const unsigned*)z, pos,
                                              (const unsigned*)batch);

  // fused edge kernel: 25000 tiles of 16 edges, 2 waves per block
  edge_moe_kernel<<<12500, 64, 0, stream>>>(edge_idx, edge_vec, z, emb, W_dt,
                                            b_dt, b_exp, t_params, ws,
                                            fws, fws + 256,
                                            out + OUT_ATOM, out + OUT_VEC);
}
